// MergeMetaCNN_14242111553932
// MI455X (gfx1250) — matrix operand order repaired and checked by reading, not compiled, not run on hardware
//
#include <hip/hip_runtime.h>
#include <hip/hip_bf16.h>

// ---------------------------------------------------------------------------
// MergeMetaCNN for MI455X (gfx1250, wave32)
//   K0: cnn_w (64x8x3x3 f32) -> Wk[64][96] f16 (K = t*9+kh*3+kw, zero pad)
//   K1: hypernet MLP -> dynamic conv weights (scaled) + bias per sample
//   K2: dynamic grouped 3x3 conv (VALU), y written as f16 (fits in L2)
//   K3: dense 3x3 conv 8->64 as implicit GEMM via v_wmma_f32_16x16x32_f16
// ---------------------------------------------------------------------------

typedef __attribute__((ext_vector_type(16))) _Float16 v16h;
typedef __attribute__((ext_vector_type(8)))  float    v8f;

#define B_     32
#define CIN_   3
#define HW_    224
#define TMP_   8
#define COUT_  64
#define FLAT_  128
#define MLPO_  224           // 8*3*3*3 + 8
#define KPAD_  96            // 72 padded to 3 chunks of 32

// ---------------- K0: pack cnn weights into f16 [64][96] -------------------
__global__ void k_pack_w(const float* __restrict__ cnn_w, _Float16* __restrict__ Wk) {
    int idx = blockIdx.x * blockDim.x + threadIdx.x;       // 64*96 = 6144
    if (idx >= COUT_ * KPAD_) return;
    int o = idx / KPAD_, k = idx - o * KPAD_;
    float v = (k < 72) ? cnn_w[o * 72 + k] : 0.0f;
    Wk[idx] = (_Float16)v;
}

// ---------------- K1: hypernetwork MLP (one block per sample) --------------
__global__ void k_mlp(const float* __restrict__ fx, const float* __restrict__ W1,
                      const float* __restrict__ b1, const float* __restrict__ W2,
                      const float* __restrict__ b2,
                      float* __restrict__ dynwt, float* __restrict__ dynb) {
    __shared__ float hid[MLPO_];
    const int b = blockIdx.x;
    const float* xr = fx + b * FLAT_;
    for (int j = threadIdx.x; j < MLPO_; j += blockDim.x) {
        float s = b1[j];
        #pragma unroll 4
        for (int k = 0; k < FLAT_; ++k) s += xr[k] * W1[k * MLPO_ + j];
        hid[j] = fmaxf(s, 0.0f);
    }
    __syncthreads();
    const float scale = 0.1f / 27.0f;                       // META_FACTOR / (cin*K*K)
    for (int j = threadIdx.x; j < MLPO_; j += blockDim.x) {
        float s = b2[j];
        #pragma unroll 4
        for (int k = 0; k < MLPO_; ++k) s += hid[k] * W2[k * MLPO_ + j];
        if (j < 216) dynwt[b * 216 + j] = s * scale;
        else         dynb [b * TMP_ + (j - 216)] = s * 0.1f;
    }
}

// ---------------- K2: dynamic grouped conv, y -> f16 -----------------------
__global__ void k_conv1(const float* __restrict__ X, const float* __restrict__ dynwt,
                        const float* __restrict__ dynb, _Float16* __restrict__ y16) {
    __shared__ float lw[216];
    __shared__ float lb[TMP_];
    const int b = blockIdx.x / 196;
    const int chunk = blockIdx.x - b * 196;
    if (threadIdx.x < 216) lw[threadIdx.x] = dynwt[b * 216 + threadIdx.x];
    if (threadIdx.x < TMP_) lb[threadIdx.x] = dynb[b * TMP_ + threadIdx.x];
    __syncthreads();

    const int pix = chunk * 256 + threadIdx.x;              // < 50176
    const int h = pix / HW_, w = pix - h * HW_;

    float acc[TMP_];
    #pragma unroll
    for (int t = 0; t < TMP_; ++t) acc[t] = lb[t];

    const float* Xb = X + (size_t)b * CIN_ * HW_ * HW_;
    __builtin_prefetch(Xb + (h + 1 < HW_ ? h + 1 : h) * HW_ + w, 0, 0);  // global_prefetch_b8

    #pragma unroll
    for (int c = 0; c < CIN_; ++c) {
        #pragma unroll
        for (int kh = 0; kh < 3; ++kh) {
            int hh = h + kh - 1;
            #pragma unroll
            for (int kw = 0; kw < 3; ++kw) {
                int ww = w + kw - 1;
                float xv = 0.0f;
                if (hh >= 0 && hh < HW_ && ww >= 0 && ww < HW_)
                    xv = Xb[(c * HW_ + hh) * HW_ + ww];
                #pragma unroll
                for (int t = 0; t < TMP_; ++t)
                    acc[t] = fmaf(xv, lw[t * 27 + c * 9 + kh * 3 + kw], acc[t]);
            }
        }
    }
    #pragma unroll
    for (int t = 0; t < TMP_; ++t)
        y16[(((size_t)b * TMP_ + t) * HW_ + h) * HW_ + w] = (_Float16)acc[t];
}

// ---------------- K3: dense conv 8->64 via WMMA implicit GEMM --------------
// One wave per (sample, row). M=64 (4 tiles), K=96 (3 steps of 32), N=14x16 px.
__global__ void __launch_bounds__(256)
k_conv2(const _Float16* __restrict__ y16, const _Float16* __restrict__ Wk,
        const float* __restrict__ cnn_b, float* __restrict__ out) {
    __shared__ _Float16 ldsWk[COUT_ * KPAD_];               // 12 KB
    {   // cooperative stage of packed weights
        const uint* src = (const uint*)Wk;
        uint* dst = (uint*)ldsWk;
        for (int i = threadIdx.x; i < (COUT_ * KPAD_) / 2; i += 256) dst[i] = src[i];
    }
    __syncthreads();

    const int lane  = threadIdx.x & 31;
    const int wave  = threadIdx.x >> 5;                     // wave32
    const int gw    = blockIdx.x * 8 + wave;                // 896*8 = 32*224
    const int b     = gw / HW_;
    const int h     = gw - b * HW_;

    const int laneM = lane & 15;
    const int hiA   = (lane >> 4) ? 8 : 0;                  // A: upper-half lanes hold K+8
    const int hiB   = (lane >> 4) ? 16 : 0;                 // (unused) B uses hiA table order

    // A fragments (weights), ISA 16-bit A 16x32 lane layout, kept in VGPRs
    v16h a[4][3];
    #pragma unroll
    for (int m = 0; m < 4; ++m) {
        const int o = m * 16 + laneM;
        #pragma unroll
        for (int s = 0; s < 3; ++s) {
            #pragma unroll
            for (int i = 0; i < 16; ++i) {
                int kl = (i < 8 ? i : i + 8) + hiA;         // K within 32-chunk
                a[m][s][i] = ldsWk[o * KPAD_ + 32 * s + kl];
            }
        }
    }

    const _Float16* yb = y16 + (size_t)b * TMP_ * HW_ * HW_;

    for (int n = 0; n < 14; ++n) {                          // 14 N-tiles of 16 px
        const int col = n * 16 + laneM;
        v8f c0 = {}, c1 = {}, c2 = {}, c3 = {};

        #pragma unroll
        for (int s = 0; s < 3; ++s) {
            // B fragment: ISA 16-bit B 32x16 lane layout (im2col gather from L2)
            v16h bf;
            #pragma unroll
            for (int i = 0; i < 16; ++i) {
                int k = 32 * s + (i < 8 ? i : i + 8) + hiA;
                _Float16 v = (_Float16)0.0f;
                if (k < 72) {
                    int t  = k / 9;
                    int r  = k - t * 9;
                    int kh = r / 3;
                    int kw = r - kh * 3;
                    int hh = h + kh - 1, ww = col + kw - 1;
                    if (hh >= 0 && hh < HW_ && ww >= 0 && ww < HW_)
                        v = yb[(t * HW_ + hh) * HW_ + ww];
                }
                bf[i] = v;
            }
            c0 = __builtin_amdgcn_wmma_f32_16x16x32_f16(false, a[0][s], false, bf, (short)0, c0, false, false);
            c1 = __builtin_amdgcn_wmma_f32_16x16x32_f16(false, a[1][s], false, bf, (short)0, c1, false, false);
            c2 = __builtin_amdgcn_wmma_f32_16x16x32_f16(false, a[2][s], false, bf, (short)0, c2, false, false);
            c3 = __builtin_amdgcn_wmma_f32_16x16x32_f16(false, a[3][s], false, bf, (short)0, c3, false, false);
        }

        // C/D layout: lane holds (cout = m*16 + r + 8*hi, pixel = lane&15)
        const int pcol = n * 16 + laneM;
        const int rof  = (lane >> 4) ? 8 : 0;
        #pragma unroll
        for (int r = 0; r < 8; ++r) {
            int o0 = 0 * 16 + r + rof, o1 = 1 * 16 + r + rof;
            int o2 = 2 * 16 + r + rof, o3 = 3 * 16 + r + rof;
            size_t base = (((size_t)b * COUT_) * HW_ + h) * HW_ + pcol;
            out[base + (size_t)o0 * HW_ * HW_] = fmaxf(c0[r] + cnn_b[o0], 0.0f);
            out[base + (size_t)o1 * HW_ * HW_] = fmaxf(c1[r] + cnn_b[o1], 0.0f);
            out[base + (size_t)o2 * HW_ * HW_] = fmaxf(c2[r] + cnn_b[o2], 0.0f);
            out[base + (size_t)o3 * HW_ * HW_] = fmaxf(c3[r] + cnn_b[o3], 0.0f);
        }
    }
}

// ---------------------------------------------------------------------------
extern "C" void kernel_launch(void* const* d_in, const int* in_sizes, int n_in,
                              void* d_out, int out_size, void* d_ws, size_t ws_size,
                              hipStream_t stream) {
    (void)in_sizes; (void)n_in; (void)out_size; (void)ws_size;
    const float* X     = (const float*)d_in[0];
    const float* fx    = (const float*)d_in[1];
    const float* W1    = (const float*)d_in[2];
    const float* b1    = (const float*)d_in[3];
    const float* W2    = (const float*)d_in[4];
    const float* b2    = (const float*)d_in[5];
    const float* cnn_w = (const float*)d_in[6];
    const float* cnn_b = (const float*)d_in[7];
    float* out = (float*)d_out;

    // workspace layout
    char* ws = (char*)d_ws;
    _Float16* y16   = (_Float16*)ws;                                    // 25,690,112 B
    float*    dynwt = (float*)(ws + 25690112);                          //     27,648 B
    float*    dynb  = (float*)(ws + 25690112 + 27648);                  //      1,024 B
    _Float16* Wk    = (_Float16*)(ws + 25690112 + 27648 + 1024);        //     12,288 B

    k_pack_w<<<24, 256, 0, stream>>>(cnn_w, Wk);
    k_mlp   <<<B_, 256, 0, stream>>>(fx, W1, b1, W2, b2, dynwt, dynb);
    k_conv1 <<<B_ * 196, 256, 0, stream>>>(X, dynwt, dynb, y16);
    k_conv2 <<<B_ * HW_ / 8, 256, 0, stream>>>(y16, Wk, cnn_b, out);
}
